// DotProductAttention_51926154609341
// MI455X (gfx1250) — matrix operand order repaired and checked by reading, not compiled, not run on hardware
//
#include <hip/hip_runtime.h>

typedef __attribute__((ext_vector_type(16))) _Float16 v16h;
typedef __attribute__((ext_vector_type(8)))  _Float16 v8h;
typedef __attribute__((ext_vector_type(8)))  float    v8f;
typedef __attribute__((ext_vector_type(4)))  float    v4f;

namespace {
constexpr int   BB    = 8;
constexpr int   SS    = 4096;
constexpr int   DD    = 64;
constexpr int   TS    = 16;
constexpr int   NT    = SS / TS;     // 256 key tiles
constexpr int   WAVES = 8;
constexpr float NEGV  = -1e9f;
constexpr float SCALE = 0.125f;      // 1/sqrt(64)
}

__device__ __forceinline__ v8f wmma_f16(v16h a, v16h b, v8f c) {
  // v_wmma_f32_16x16x32_f16: (neg_a, A, neg_b, B, c_mod, C, reuse_a, reuse_b)
  return __builtin_amdgcn_wmma_f32_16x16x32_f16(false, a, false, b, (short)0, c, false, false);
}

// A-fragment (16x32 f16) of one K tile, chunk c (d = 32c..32c+31).
// Lane(0-15) half j -> d = 32c + j + (j>=8?8:0); lanes 16-31: +8.
template <bool USE_WS>
__device__ __forceinline__ v16h k_frag(const float* __restrict__ krow,
                                       const _Float16* __restrict__ krowh, int c, int h) {
  v16h a;
  if constexpr (USE_WS) {
    v8h x = *(const v8h*)(krowh + 32 * c + 8 * h);
    v8h y = *(const v8h*)(krowh + 32 * c + 16 + 8 * h);
#pragma unroll
    for (int j = 0; j < 8; ++j) { a[j] = x[j]; a[j + 8] = y[j]; }
  } else {
    const float* p0 = krow + 32 * c + 8 * h;
    const float* p1 = krow + 32 * c + 16 + 8 * h;
    v4f x0 = *(const v4f*)(p0);
    v4f x1 = *(const v4f*)(p0 + 4);
    v4f y0 = *(const v4f*)(p1);
    v4f y1 = *(const v4f*)(p1 + 4);
#pragma unroll
    for (int j = 0; j < 4; ++j) {
      a[j]      = (_Float16)x0[j];
      a[j + 4]  = (_Float16)x1[j];
      a[j + 8]  = (_Float16)y0[j];
      a[j + 12] = (_Float16)y1[j];
    }
  }
  return a;
}

// S^T tile = K_tile x Q^T  (D-frag: lane holds query row m=lane%16, 8 key cols)
template <bool USE_WS>
__device__ __forceinline__ v8f score_tile(const float* __restrict__ Kb,
                                          const _Float16* __restrict__ Khb,
                                          int kt, int lm, int h, v16h qb0, v16h qb1) {
  const float*    krow  = Kb  + (size_t)(kt * TS + lm) * DD;
  const _Float16* krowh = Khb + (size_t)(kt * TS + lm) * DD;
  v8f c = {};
  c = wmma_f16(k_frag<USE_WS>(krow, krowh, 0, h), qb0, c);
  c = wmma_f16(k_frag<USE_WS>(krow, krowh, 1, h), qb1, c);
  return c;
}

// B-fragment (32x16) of V: lane column d = 16f + lane%16, halves j -> k = kt2*32 + 8h + j + (j>=8?8:0)
template <bool USE_WS>
__device__ __forceinline__ v16h v_frag(const float* __restrict__ Vb,
                                       const _Float16* __restrict__ vt,
                                       int b, int f, int lm, int h, int kt2) {
  v16h bv;
  const int kbase = kt2 * 32 + 8 * h;
  const int d     = f * 16 + lm;
  if constexpr (USE_WS) {
    const _Float16* p = vt + ((size_t)(b * DD + d)) * SS + kbase;
    v8h lo = *(const v8h*)(p);
    v8h hi = *(const v8h*)(p + 16);
#pragma unroll
    for (int j = 0; j < 8; ++j) { bv[j] = lo[j]; bv[j + 8] = hi[j]; }
  } else {
    const float* p = Vb + (size_t)kbase * DD + d;
#pragma unroll
    for (int j = 0; j < 16; ++j) bv[j] = (_Float16)p[(size_t)(j + (j & 8)) * DD];
  }
  return bv;
}

// K[b][n][d] f32 -> f16 (same layout); V[b][k][d] f32 -> Vt[b][d][k] f16
__global__ void prep_kernel(const float* __restrict__ k, const float* __restrict__ v,
                            _Float16* __restrict__ kh, _Float16* __restrict__ vt, int n) {
  int idx = blockIdx.x * 256 + threadIdx.x;
  if (idx >= n) return;
  kh[idx] = (_Float16)k[idx];
  int kk = idx % SS;
  int d  = (idx / SS) % DD;
  int b  = idx / (SS * DD);
  vt[idx] = (_Float16)v[((size_t)b * SS + kk) * DD + d];
}

template <bool USE_WS>
__global__ __launch_bounds__(WAVES * 32)
void attn_kernel(const float* __restrict__ q, const float* __restrict__ k,
                 const float* __restrict__ v, const int* __restrict__ lens,
                 const _Float16* __restrict__ kh, const _Float16* __restrict__ vt,
                 float* __restrict__ out, float* __restrict__ attn) {
  const int lane = threadIdx.x & 31;
  const int wave = threadIdx.x >> 5;
  const int tile = blockIdx.x * WAVES + wave;    // 0 .. B*S/16-1
  const int b    = tile / (SS / TS);
  const int m0   = (tile % (SS / TS)) * TS;
  const int lm   = lane & 15;                    // query row within tile
  const int h    = lane >> 4;                    // key-column half
  const int len  = lens[b];                      // wave-uniform

  const float*    Qb  = q + (size_t)b * SS * DD;
  const float*    Kb  = k + (size_t)b * SS * DD;
  const float*    Vb  = v + (size_t)b * SS * DD;
  const _Float16* Khb = kh + (size_t)b * SS * DD;
  float* attnB        = attn + (size_t)b * SS * SS;

  // ---- Q B-fragments (Q^T chunks, pre-scaled by 1/sqrt(D)), loaded once ----
  v16h qb0, qb1;
  {
    const float* qrow = Qb + (size_t)(m0 + lm) * DD + 8 * h;
#pragma unroll
    for (int j = 0; j < 16; j += 4) {
      v4f a  = *(const v4f*)(qrow + j + (j & 8));
      v4f bq = *(const v4f*)(qrow + 32 + j + (j & 8));
#pragma unroll
      for (int t = 0; t < 4; ++t) {
        qb0[j + t] = (_Float16)(a[t] * SCALE);
        qb1[j + t] = (_Float16)(bq[t] * SCALE);
      }
    }
  }

  // ---- Pass 1: per-lane online max/exp-sum, valid prefix only ----
  const int ktFull = len >> 4;                   // fully valid 16-wide tiles
  float M = -1e30f, L = 0.f;
  for (int kt = 0; kt < ktFull; ++kt) {
    if constexpr (USE_WS) __builtin_prefetch(Khb + (size_t)((kt + 1) * TS + lm) * DD, 0, 0);
    v8f c = score_tile<USE_WS>(Kb, Khb, kt, lm, h, qb0, qb1);
    float t0 = fmaxf(fmaxf(c[0], c[1]), fmaxf(c[2], c[3]));
    float t1 = fmaxf(fmaxf(c[4], c[5]), fmaxf(c[6], c[7]));
    float Mn = fmaxf(M, fmaxf(t0, t1));
    L *= __expf(M - Mn);
    M = Mn;
#pragma unroll
    for (int r = 0; r < 8; ++r) L += __expf(c[r] - M);
  }
  if (len & 15) {                                // boundary tile (uniform branch)
    const int kt = ktFull;
    v8f c = score_tile<USE_WS>(Kb, Khb, kt, lm, h, qb0, qb1);
    const int nb = kt * TS + 8 * h;
    float tmax = -1e30f;
#pragma unroll
    for (int r = 0; r < 8; ++r) {
      float s = (nb + r < len) ? c[r] : NEGV;
      c[r] = s;
      tmax = fmaxf(tmax, s);
    }
    float Mn = fmaxf(M, tmax);
    L *= __expf(M - Mn);
    M = Mn;
#pragma unroll
    for (int r = 0; r < 8; ++r) L += __expf(c[r] - M);
  }
  // combine the two lanes (h=0/1) sharing query row m
  {
    float Mo = __shfl_xor(M, 16, 32);
    float Lo = __shfl_xor(L, 16, 32);
    float Mr = fmaxf(M, Mo);
    L = L * __expf(M - Mr) + Lo * __expf(Mo - Mr);
    M = Mr;
  }
  const float rinv = 1.0f / L;

  // ---- Pass 2: normalized attn (NT stores) + P.V over valid prefix ----
  v8f o[4] = {};
  float* arow = attnB + (size_t)(m0 + lm) * SS;

  auto do_pair = [&](v8f c0, v8f c1, int kt2) {   // c0/c1 already masked scores
    v16h ap;                                      // A-fragment of P (16x32)
    const int nb0 = kt2 * 32 + 8 * h;
    const int nb1 = kt2 * 32 + 16 + 8 * h;
#pragma unroll
    for (int r = 0; r < 8; ++r) {
      float pe0 = __expf(c0[r] - M) * rinv;
      float pe1 = __expf(c1[r] - M) * rinv;
      ap[r]     = (_Float16)pe0;
      ap[r + 8] = (_Float16)pe1;
      c0[r] = pe0;
      c1[r] = pe1;
    }
    v4f s00 = {c0[0], c0[1], c0[2], c0[3]};
    v4f s01 = {c0[4], c0[5], c0[6], c0[7]};
    v4f s10 = {c1[0], c1[1], c1[2], c1[3]};
    v4f s11 = {c1[4], c1[5], c1[6], c1[7]};
    __builtin_nontemporal_store(s00, (v4f*)(arow + nb0));
    __builtin_nontemporal_store(s01, (v4f*)(arow + nb0 + 4));
    __builtin_nontemporal_store(s10, (v4f*)(arow + nb1));
    __builtin_nontemporal_store(s11, (v4f*)(arow + nb1 + 4));
#pragma unroll
    for (int f = 0; f < 4; ++f) {
      o[f] = wmma_f16(ap, v_frag<USE_WS>(Vb, vt, b, f, lm, h, kt2), o[f]);
    }
  };

  const int  p2Full = len >> 5;                   // fully valid 32-wide pairs
  const bool p2B    = (len & 31) != 0;
  for (int kt2 = 0; kt2 < p2Full; ++kt2) {        // unmasked fast path
    v8f c0 = score_tile<USE_WS>(Kb, Khb, 2 * kt2,     lm, h, qb0, qb1);
    v8f c1 = score_tile<USE_WS>(Kb, Khb, 2 * kt2 + 1, lm, h, qb0, qb1);
    do_pair(c0, c1, kt2);
  }
  if (p2B) {                                      // masked boundary pair (uniform)
    const int kt2 = p2Full;
    v8f c0 = score_tile<USE_WS>(Kb, Khb, 2 * kt2,     lm, h, qb0, qb1);
    v8f c1 = score_tile<USE_WS>(Kb, Khb, 2 * kt2 + 1, lm, h, qb0, qb1);
    const int nb0 = kt2 * 32 + 8 * h;
    const int nb1 = kt2 * 32 + 16 + 8 * h;
#pragma unroll
    for (int r = 0; r < 8; ++r) {
      c0[r] = (nb0 + r < len) ? c0[r] : NEGV;
      c1[r] = (nb1 + r < len) ? c1[r] : NEGV;
    }
    do_pair(c0, c1, kt2);
  }
  // masked tail of each attn row is exactly zero: stream zeros, no compute
  {
    const int nz0 = (p2Full + (p2B ? 1 : 0)) * 32;
    const v4f z = {0.f, 0.f, 0.f, 0.f};
    for (int n = nz0 + 8 * h; n < SS; n += 16) {
      __builtin_nontemporal_store(z, (v4f*)(arow + n));
      __builtin_nontemporal_store(z, (v4f*)(arow + n + 4));
    }
  }

  // ---- store out[b][m0+m][d] ----
  float* ob = out + ((size_t)b * SS + m0) * DD;
#pragma unroll
  for (int f = 0; f < 4; ++f) {
#pragma unroll
    for (int r = 0; r < 8; ++r) {
      ob[(size_t)(r + 8 * h) * DD + f * 16 + lm] = o[f][r];
    }
  }
}

extern "C" void kernel_launch(void* const* d_in, const int* in_sizes, int n_in,
                              void* d_out, int out_size, void* d_ws, size_t ws_size,
                              hipStream_t stream) {
  (void)in_sizes; (void)n_in; (void)out_size;
  const float* q    = (const float*)d_in[0];
  const float* k    = (const float*)d_in[1];
  const float* v    = (const float*)d_in[2];
  const int*   lens = (const int*)d_in[3];
  float* out  = (float*)d_out;
  float* attn = out + (size_t)BB * SS * DD;

  const int    nTiles   = BB * (SS / TS);              // 2048 query tiles
  const dim3   grid(nTiles / WAVES), block(WAVES * 32);
  const size_t elems    = (size_t)BB * SS * DD;
  const size_t ws_need  = elems * 2 * 2;               // Kh (4MB) + Vt (4MB)

  if (ws_size >= ws_need) {
    _Float16* kh = (_Float16*)d_ws;
    _Float16* vt = kh + elems;
    const int n = (int)elems;
    prep_kernel<<<(n + 255) / 256, 256, 0, stream>>>(k, v, kh, vt, n);
    attn_kernel<true><<<grid, block, 0, stream>>>(q, k, v, lens, kh, vt, out, attn);
  } else {
    attn_kernel<false><<<grid, block, 0, stream>>>(q, k, v, lens, nullptr, nullptr, out, attn);
  }
}
